// Decoder_644245094507
// MI455X (gfx1250) — matrix operand order repaired and checked by reading, not compiled, not run on hardware
//
#include <hip/hip_runtime.h>

typedef __bf16 bf16_t;
typedef __attribute__((ext_vector_type(16))) __bf16 v16bf;
typedef __attribute__((ext_vector_type(8)))  float  v8f;

#define HD     512
#define BATCH  2048
#define NCH    32
#define NNOTES 8
#define CPL    64
#define MPL    128

// ---------------------------------------------------------------------------
// Fragment loaders (wave32): contiguous k order, k = 16*(lane/16) + s, same for A and B.
// All activations & weights are bf16 in memory.
// ---------------------------------------------------------------------------

// A: 16x32 bf16 (MxK), source row-major [M, lda]. lane holds M = lane%16.
// halves 0-15 -> K = k0 + 16*(lane/16) + 0..15 (32B contiguous)
__device__ __forceinline__ v16bf load_a_frag(const bf16_t* __restrict__ A, int lda,
                                             int row0, int k0, int lane) {
  const bf16_t* p = A + (size_t)(row0 + (lane & 15)) * lda + k0 + 16 * (lane >> 4);
  union { v16bf v; float4 f4[2]; } u;
  u.f4[0] = *reinterpret_cast<const float4*>(p);
  u.f4[1] = *reinterpret_cast<const float4*>(p + 8);
  return u.v;
}

// B: 32x16 bf16 (KxN). Weights stored [N, K] row-major (torch Linear layout),
// lane holds column N = col0 + lane%16; halves h -> K = k0 + 16*(lane/16) + h
// => 16 contiguous bf16 per lane (two 16B loads).
__device__ __forceinline__ v16bf load_b_frag(const bf16_t* __restrict__ W, int ldw,
                                             int col0, int k0, int N, int lane) {
  int col = col0 + (lane & 15);
  col = (col < N) ? col : (N - 1);  // clamp; clamped columns never stored
  const float4* q = reinterpret_cast<const float4*>(
      W + (size_t)col * ldw + k0 + 16 * (lane >> 4));
  union { float4 f4[2]; v16bf v; } u;
  u.f4[0] = q[0];
  u.f4[1] = q[1];
  return u.v;
}

// ---------------------------------------------------------------------------
// Generic WMMA GEMM: C[M,N] = concat_s(A_s) @ concat_s(W_s)^T + bias
// Block: 256 threads = 8 waves; wave tile 32(M) x 64(N); block tile 64 x 256.
// NSEG is a template arg so the segment loop fully unrolls and every pointer
// derives directly from a kernel argument (=> global, not flat, addressing).
// outmode: 0 = f32 store, 1 = f32+relu, 2 = bf16 store, 3 = bf16+relu
// ---------------------------------------------------------------------------
template <int NSEG>
__global__ __launch_bounds__(256)
void gemm_wmma_bf16(const bf16_t* __restrict__ A0, const bf16_t* __restrict__ A1,
                    const bf16_t* __restrict__ A2,
                    const bf16_t* __restrict__ W0, const bf16_t* __restrict__ W1,
                    const bf16_t* __restrict__ W2,
                    int ldw, int kseg,
                    const float* __restrict__ bias0, const float* __restrict__ bias1,
                    void* __restrict__ Cout, long long ldc, int N, int outmode) {
  const int lane  = threadIdx.x & 31;
  const int wave  = threadIdx.x >> 5;
  const int waveM = wave & 1;
  const int waveN = wave >> 1;
  const int row0    = blockIdx.x * 64 + waveM * 32;
  const int colBase = blockIdx.y * 256 + waveN * 64;

  v8f acc[2][4];
#pragma unroll
  for (int i = 0; i < 2; ++i)
#pragma unroll
    for (int j = 0; j < 4; ++j) { v8f zz = {}; acc[i][j] = zz; }

#pragma unroll
  for (int s = 0; s < NSEG; ++s) {
    const bf16_t* A = (s == 0) ? A0 : ((s == 1) ? A1 : A2);
    const bf16_t* W = (s == 0) ? W0 : ((s == 1) ? W1 : W2);
    for (int kt = 0; kt < kseg; kt += 32) {
      // speculative prefetch of the next weight K-tile (OOB is dropped)
      __builtin_prefetch(W + (size_t)(colBase + (lane & 15)) * ldw + kt + 32, 0, 1);
      v16bf a0 = load_a_frag(A, kseg, row0, kt, lane);
      v16bf a1 = load_a_frag(A, kseg, row0 + 16, kt, lane);
      v16bf b0 = load_b_frag(W, ldw, colBase +  0, kt, N, lane);
      v16bf b1 = load_b_frag(W, ldw, colBase + 16, kt, N, lane);
      v16bf b2 = load_b_frag(W, ldw, colBase + 32, kt, N, lane);
      v16bf b3 = load_b_frag(W, ldw, colBase + 48, kt, N, lane);
      acc[0][0] = __builtin_amdgcn_wmma_f32_16x16x32_bf16(false, a0, false, b0, (short)0, acc[0][0], false, false);
      acc[1][0] = __builtin_amdgcn_wmma_f32_16x16x32_bf16(false, a1, false, b0, (short)0, acc[1][0], false, false);
      acc[0][1] = __builtin_amdgcn_wmma_f32_16x16x32_bf16(false, a0, false, b1, (short)0, acc[0][1], false, false);
      acc[1][1] = __builtin_amdgcn_wmma_f32_16x16x32_bf16(false, a1, false, b1, (short)0, acc[1][1], false, false);
      acc[0][2] = __builtin_amdgcn_wmma_f32_16x16x32_bf16(false, a0, false, b2, (short)0, acc[0][2], false, false);
      acc[1][2] = __builtin_amdgcn_wmma_f32_16x16x32_bf16(false, a1, false, b2, (short)0, acc[1][2], false, false);
      acc[0][3] = __builtin_amdgcn_wmma_f32_16x16x32_bf16(false, a0, false, b3, (short)0, acc[0][3], false, false);
      acc[1][3] = __builtin_amdgcn_wmma_f32_16x16x32_bf16(false, a1, false, b3, (short)0, acc[1][3], false, false);
    }
  }

  // C/D layout: lane l, vgpr v -> row = v + 8*(l/16), col = l%16
  const bool as_bf16 = (outmode >= 2);
  const bool do_relu = (outmode & 1);
#pragma unroll
  for (int i = 0; i < 2; ++i) {
    int r0 = row0 + i * 16 + 8 * (lane >> 4);
#pragma unroll
    for (int j = 0; j < 4; ++j) {
      int col = colBase + 16 * j + (lane & 15);
      if (col < N) {
        float bsum = (bias0 ? bias0[col] : 0.f) + (bias1 ? bias1[col] : 0.f);
        if (as_bf16) {
          bf16_t* C = (bf16_t*)Cout;
#pragma unroll
          for (int v = 0; v < 8; ++v) {
            float val = acc[i][j][v] + bsum;
            if (do_relu) val = fmaxf(val, 0.f);
            C[(size_t)(r0 + v) * ldc + col] = (bf16_t)val;
          }
        } else {
          float* C = (float*)Cout;
#pragma unroll
          for (int v = 0; v < 8; ++v) {
            float val = acc[i][j][v] + bsum;
            if (do_relu) val = fmaxf(val, 0.f);
            C[(size_t)(r0 + v) * ldc + col] = val;
          }
        }
      }
    }
  }
}

// ---------------------------------------------------------------------------
// Elementwise / small kernels
// ---------------------------------------------------------------------------
// c stays fp32 (carried state); h emitted as bf16 (only ever a WMMA A-operand)
__global__ void lstm_pointwise(const float* __restrict__ g,
                               const float* __restrict__ c_in,
                               float* __restrict__ c_out,
                               bf16_t* __restrict__ h_out) {
  int idx = blockIdx.x * blockDim.x + threadIdx.x;  // BATCH*HD
  int b = idx >> 9, j = idx & (HD - 1);
  const float* gr = g + (size_t)b * (4 * HD);
  float i_ = gr[j], f_ = gr[HD + j], gg = gr[2 * HD + j], o_ = gr[3 * HD + j];
  float si = 1.f / (1.f + __expf(-i_));
  float sf = 1.f / (1.f + __expf(-f_));
  float so = 1.f / (1.f + __expf(-o_));
  float c  = sf * c_in[idx] + si * tanhf(gg);
  c_out[idx] = c;
  h_out[idx] = (bf16_t)(so * tanhf(c));
}

// warp-per-row argmax (first-max tiebreak like jnp.argmax) + bf16 embedding gather
__global__ void argmax_embed(const float* __restrict__ logits, long long ldl,
                             int ncols, const bf16_t* __restrict__ emb,
                             bf16_t* __restrict__ out) {
  int row  = (blockIdx.x * blockDim.x + threadIdx.x) >> 5;
  int lane = threadIdx.x & 31;
  if (row >= BATCH) return;
  const float* lr = logits + (size_t)row * ldl;
  float best = -3.4e38f; int bidx = 0x7fffffff;
  for (int c = lane; c < ncols; c += 32) {
    float v = lr[c];
    if (v > best) { best = v; bidx = c; }
  }
  for (int off = 16; off > 0; off >>= 1) {
    float ov = __shfl_xor(best, off, 32);
    int   oi = __shfl_xor(bidx, off, 32);
    if (ov > best || (ov == best && oi < bidx)) { best = ov; bidx = oi; }
  }
  const unsigned* er = reinterpret_cast<const unsigned*>(emb + (size_t)bidx * HD);
  unsigned* orow = reinterpret_cast<unsigned*>(out + (size_t)row * HD);
  for (int j = lane; j < HD / 2; j += 32) orow[j] = er[j];
}

__global__ void vec_add_bf16(const bf16_t* a, const bf16_t* b, bf16_t* c, int n) {
  int i = blockIdx.x * blockDim.x + threadIdx.x;
  if (i < n) c[i] = (bf16_t)((float)a[i] + (float)b[i]);
}
__global__ void vec_zero_bf16(bf16_t* a, int n) {
  int i = blockIdx.x * blockDim.x + threadIdx.x;
  if (i < n) a[i] = (bf16_t)0.f;
}
__global__ void cvt_bf16_kernel(const float* a, bf16_t* o, int n) {
  int i = blockIdx.x * blockDim.x + threadIdx.x;
  if (i < n) o[i] = (bf16_t)a[i];
}

// ---------------------------------------------------------------------------
extern "C" void kernel_launch(void* const* d_in, const int* in_sizes, int n_in,
                              void* d_out, int out_size, void* d_ws, size_t ws_size,
                              hipStream_t stream) {
  (void)in_sizes; (void)n_in; (void)out_size; (void)ws_size;
  auto F = [&](int i) { return (const float*)d_in[i]; };

  const float *z = F(0), *hxc = F(1), *cxc = F(2), *hxm = F(3), *cxm = F(4);
  const float *cWih = F(5), *cWhh = F(6), *c_bih = F(7), *c_bhh = F(8), *c_emb = F(9);
  const float *cp1W = F(10), *cp1b = F(11), *cp2W = F(12), *cp2b = F(13);
  const float *cdsW = F(14), *cdsb = F(15);
  const float *mWih = F(16), *mWhh = F(17), *m_bih = F(18), *m_bhh = F(19), *m_emb = F(20);
  const float *mp1W = F(21), *mp1b = F(22), *mp2W = F(23), *mp2b = F(24);
  const float *mdsW = F(25), *mdsb = F(26);
  // heads start at 27: (W1,b1,W2,b2) x {tempo,key,mode,valence,energy}

  // ---- workspace layout ----
  size_t off = 0;
  auto alloc = [&](size_t bytes) {
    off = (off + 255) & ~(size_t)255;
    size_t o = off; off += bytes; return o;
  };
  char* ws = (char*)d_ws;
  auto BW = [&](size_t nelem) { return (bf16_t*)(ws + alloc(nelem * 2)); };
  auto FW = [&](size_t nelem) { return (float*)(ws + alloc(nelem * 4)); };

  // bf16 weights
  bf16_t* wcWih = BW(4 * HD * HD); bf16_t* wcWhh = BW(4 * HD * HD);
  bf16_t* wmWih = BW(4 * HD * HD); bf16_t* wmWhh = BW(4 * HD * HD);
  bf16_t* wcp1 = BW(HD * HD);  bf16_t* wcp2 = BW(CPL * HD);
  bf16_t* wmp1 = BW(HD * HD);  bf16_t* wmp2 = BW(MPL * HD);
  bf16_t* wcds = BW(HD * 2 * HD); bf16_t* wmds = BW(HD * 3 * HD);
  bf16_t* wh1[5]; bf16_t* wh2[5];
  const int hout[5] = {1, 12, 7, 1, 1};
  for (int h = 0; h < 5; ++h) { wh1[h] = BW(256 * HD); wh2[h] = BW((size_t)hout[h] * 256); }
  // bf16 activations / embeddings
  bf16_t* zb   = BW((size_t)BATCH * HD);
  bf16_t* cembb = BW((size_t)CPL * HD);
  bf16_t* membb = BW((size_t)MPL * HD);
  bf16_t* hc = BW((size_t)BATCH * HD); bf16_t* hm = BW((size_t)BATCH * HD);
  bf16_t* tbuf = BW((size_t)BATCH * HD);
  bf16_t* ce0 = BW((size_t)BATCH * HD); bf16_t* ce = BW((size_t)BATCH * HD);
  bf16_t* me0 = BW((size_t)BATCH * HD); bf16_t* me = BW((size_t)BATCH * HD);
  bf16_t* melin = BW((size_t)BATCH * HD);
  bf16_t* headh = BW((size_t)BATCH * 256);
  // fp32 buffers
  float* gates = FW((size_t)BATCH * 4 * HD);
  float* cc = FW((size_t)BATCH * HD); float* cm = FW((size_t)BATCH * HD);

  auto cvt = [&](const float* src, bf16_t* dst, size_t n) {
    cvt_bf16_kernel<<<(unsigned)((n + 255) / 256), 256, 0, stream>>>(src, dst, (int)n);
  };
  cvt(cWih, wcWih, 4 * HD * HD); cvt(cWhh, wcWhh, 4 * HD * HD);
  cvt(mWih, wmWih, 4 * HD * HD); cvt(mWhh, wmWhh, 4 * HD * HD);
  cvt(cp1W, wcp1, HD * HD); cvt(cp2W, wcp2, CPL * HD);
  cvt(mp1W, wmp1, HD * HD); cvt(mp2W, wmp2, MPL * HD);
  cvt(cdsW, wcds, HD * 2 * HD); cvt(mdsW, wmds, HD * 3 * HD);
  for (int h = 0; h < 5; ++h) {
    cvt(F(27 + 4 * h + 0), wh1[h], 256 * HD);
    cvt(F(27 + 4 * h + 2), wh2[h], (size_t)hout[h] * 256);
  }
  cvt(z, zb, (size_t)BATCH * HD);
  cvt(c_emb, cembb, (size_t)CPL * HD);
  cvt(m_emb, membb, (size_t)MPL * HD);
  cvt(hxc, hc, (size_t)BATCH * HD);
  cvt(hxm, hm, (size_t)BATCH * HD);
  hipMemcpyAsync(cc, cxc, (size_t)BATCH * HD * sizeof(float), hipMemcpyDeviceToDevice, stream);
  hipMemcpyAsync(cm, cxm, (size_t)BATCH * HD * sizeof(float), hipMemcpyDeviceToDevice, stream);
  vec_zero_bf16<<<(BATCH * HD) / 256, 256, 0, stream>>>(me, BATCH * HD);

  auto gemm = [&](int nseg, const bf16_t* a0, const bf16_t* w0, const bf16_t* a1,
                  const bf16_t* w1, const bf16_t* a2, const bf16_t* w2, int kseg,
                  int ldw, const float* b0, const float* b1, void* C, long long ldc,
                  int N, int outmode) {
    dim3 grid(BATCH / 64, (unsigned)((N + 255) / 256));
    switch (nseg) {
      case 1: gemm_wmma_bf16<1><<<grid, 256, 0, stream>>>(a0, a1, a2, w0, w1, w2, ldw, kseg, b0, b1, C, ldc, N, outmode); break;
      case 2: gemm_wmma_bf16<2><<<grid, 256, 0, stream>>>(a0, a1, a2, w0, w1, w2, ldw, kseg, b0, b1, C, ldc, N, outmode); break;
      default: gemm_wmma_bf16<3><<<grid, 256, 0, stream>>>(a0, a1, a2, w0, w1, w2, ldw, kseg, b0, b1, C, ldc, N, outmode); break;
    }
  };

  float* out = (float*)d_out;
  const size_t NOTE_OFF = (size_t)BATCH * NCH * CPL;
  const size_t HEAD_OFF = NOTE_OFF + (size_t)BATCH * NCH * NNOTES * MPL;
  size_t hoff[5];
  { size_t o = HEAD_OFF;
    for (int h = 0; h < 5; ++h) { hoff[h] = o; o += (size_t)BATCH * hout[h]; } }

  // ---- heads (independent of the scan) ----
  for (int h = 0; h < 5; ++h) {
    gemm(1, zb, wh1[h], nullptr, nullptr, nullptr, nullptr, HD, HD,
         F(27 + 4 * h + 1), nullptr, headh, 256, 256, /*bf16+relu*/ 3);
    gemm(1, headh, wh2[h], nullptr, nullptr, nullptr, nullptr, 256, 256,
         F(27 + 4 * h + 3), nullptr, out + hoff[h], hout[h], hout[h], /*f32*/ 0);
  }

  // ---- sequential chord / note scan (num_chords == 32 in reference) ----
  for (int s = 0; s < NCH; ++s) {
    const bf16_t* cin = (s == 0) ? zb : ce;
    // chord LSTM gates: cin@Wih^T + hc@Whh^T + bih + bhh  -> f32 gates
    gemm(2, cin, wcWih, hc, wcWhh, nullptr, nullptr, HD, HD,
         c_bih, c_bhh, gates, 4 * HD, 4 * HD, 0);
    lstm_pointwise<<<(BATCH * HD) / 256, 256, 0, stream>>>(gates, cc, cc, hc);
    gemm(1, hc, wcp1, nullptr, nullptr, nullptr, nullptr, HD, HD,
         cp1b, nullptr, tbuf, HD, HD, /*bf16+relu*/ 3);
    // chord logits -> d_out[b, s, :]  (f32)
    gemm(1, tbuf, wcp2, nullptr, nullptr, nullptr, nullptr, HD, HD,
         cp2b, nullptr, out + (size_t)s * CPL, (long long)NCH * CPL, CPL, 0);
    argmax_embed<<<BATCH / 8, 256, 0, stream>>>(out + (size_t)s * CPL,
                                                (long long)NCH * CPL, CPL, cembb, ce0);
    // ce = concat([ce0, z]) @ c_ds_W^T + c_ds_b   (segmented K) -> bf16
    gemm(2, ce0, wcds, zb, wcds + HD, nullptr, nullptr, HD, 2 * HD,
         cdsb, nullptr, ce, HD, HD, 2);
    vec_add_bf16<<<(BATCH * HD) / 256, 256, 0, stream>>>(me, ce, melin, BATCH * HD);

    for (int n = 0; n < NNOTES; ++n) {
      const bf16_t* min_ = (n == 0) ? melin : me;
      gemm(2, min_, wmWih, hm, wmWhh, nullptr, nullptr, HD, HD,
           m_bih, m_bhh, gates, 4 * HD, 4 * HD, 0);
      lstm_pointwise<<<(BATCH * HD) / 256, 256, 0, stream>>>(gates, cm, cm, hm);
      gemm(1, hm, wmp1, nullptr, nullptr, nullptr, nullptr, HD, HD,
           mp1b, nullptr, tbuf, HD, HD, 3);
      float* nlbase = out + NOTE_OFF + (size_t)(s * NNOTES + n) * MPL;
      gemm(1, tbuf, wmp2, nullptr, nullptr, nullptr, nullptr, HD, HD,
           mp2b, nullptr, nlbase, (long long)NCH * NNOTES * MPL, MPL, 0);
      argmax_embed<<<BATCH / 8, 256, 0, stream>>>(nlbase, (long long)NCH * NNOTES * MPL,
                                                  MPL, membb, me0);
      // me = concat([me0, ce, z]) @ m_ds_W^T + m_ds_b -> bf16
      gemm(3, me0, wmds, ce, wmds + HD, zb, wmds + 2 * HD, HD, 3 * HD,
           mdsb, nullptr, me, HD, HD, 2);
    }
  }
}
